// MultiScaleWarpLayer_90787018703697
// MI455X (gfx1250) — matrix operand order repaired and checked by reading, not compiled, not run on hardware
//
#include <hip/hip_runtime.h>
#include <stdint.h>

#define NB      2
#define DMODEL  256
#define DFLOW   128
#define DFFN    1024
#define NHEADS  8
#define NLEV    3
#define NPTS    4
#define DHEAD   32
#define LQ      16384
#define LENIN   21504

typedef __attribute__((ext_vector_type(16))) __bf16 v16bf;
typedef __attribute__((ext_vector_type(8)))  float  v8f;
typedef __attribute__((ext_vector_type(4)))  unsigned int u32x4;
typedef __attribute__((ext_vector_type(8)))  int i32x8;
typedef __attribute__((ext_vector_type(4)))  int i32x4;

__device__ __forceinline__ uint16_t f2bf(float f) {
  union { float f; uint32_t u; } c; c.f = f;
  return (uint16_t)((c.u + 0x7FFFu + ((c.u >> 16) & 1u)) >> 16);
}

// ---------------- conversion kernels ----------------
__global__ void msw_f32_to_bf16(const float* __restrict__ in, uint16_t* __restrict__ out, int n) {
  int i = blockIdx.x * blockDim.x + threadIdx.x;
  if (i < n) out[i] = f2bf(in[i]);
}

// W [K,Nd] f32 (row-major) -> Wt [Nd,K] bf16
__global__ void msw_transpose_bf16(const float* __restrict__ W, uint16_t* __restrict__ Wt, int K, int Nd) {
  int i = blockIdx.x * blockDim.x + threadIdx.x;
  if (i >= K * Nd) return;
  int k = i / Nd, n = i % Nd;
  Wt[(size_t)n * K + k] = f2bf(W[i]);
}

// ---------------- bf16 WMMA GEMM, TDM-staged LDS B slab ----------------
// Block = 256 threads = 8 waves computing a 256-row x 32-col C tile.
// The 32-col weight slab (contiguous, since Bt is [Nd][K]) is DMA'd into LDS
// by the Tensor Data Mover (one wave issues TENSOR_LOAD_TO_LDS, waits
// TENSORcnt, block barriers). Each wave computes 32x32 (2x2 WMMA tiles):
// 2 A fragments x 2 B fragments -> 4 independent accumulator chains.
// Requires: M % 256 == 0, Nd % 32 == 0, K % 32 == 0 (true for all layer GEMMs).
__global__ void msw_gemm_bf16(const uint16_t* __restrict__ A,
                              const uint16_t* __restrict__ Bt,
                              const float* __restrict__ bias,
                              const float* __restrict__ resid,
                              float* __restrict__ outF,
                              uint16_t* __restrict__ outBf,
                              int M, int K, int Nd, int relu) {
  extern __shared__ uint16_t ldsB[];                 // [32][K] bf16 (col-major slabs)

  const int nb_n  = Nd >> 5;
  const int bm    = blockIdx.x / nb_n;
  const int bn    = blockIdx.x % nb_n;
  const int m_blk = bm << 8;                         // 256 rows per block
  const int n_blk = bn << 5;                         // 32 cols per block

  // --- TDM: DMA the contiguous 32*K bf16 weight slab into LDS -------------
  if (threadIdx.x < 32) {                            // one wave issues the DMA
    const uint64_t gaddr  = (uint64_t)(uintptr_t)(Bt + (size_t)n_blk * K);
    const uint32_t laddr  = (uint32_t)(uintptr_t)ldsB;
    const uint32_t nelem8 = (uint32_t)(32u * (uint32_t)K * 2u) >> 3;  // 8-byte units (<= 8192)

    u32x4 g0; i32x8 g1; i32x4 g2, g3; i32x8 g4;
    // D# group 0: count=1 user descriptor | lds_addr | global_addr | type=2
    g0[0] = 1u;
    g0[1] = laddr;
    g0[2] = (uint32_t)(gaddr & 0xFFFFFFFFull);
    g0[3] = (uint32_t)((gaddr >> 32) & 0x1FFFFFFull) | (2u << 30);
    // D# group 1: no multicast, data_size=8B(code 3), no barrier/iterate/pad
    g1[0] = (int)(3u << 16);
    g1[1] = (int)((nelem8 & 0xFFFFu) << 16);         // tensor_dim0[15:0]
    g1[2] = (int)((nelem8 >> 16) | (1u << 16));      // tensor_dim0[31:16] | tensor_dim1=1
    g1[3] = (int)(nelem8 << 16);                     // tile_dim0 = nelem8
    g1[4] = 0;                                       // tile_dim1/2 unused
    g1[5] = (int)nelem8;                             // tensor_dim0_stride[31:0]
    g1[6] = 0; g1[7] = 0;
    g2[0] = g2[1] = g2[2] = g2[3] = 0;               // 2-D tensor: groups 2/3 unused
    g3[0] = g3[1] = g3[2] = g3[3] = 0;
    g4[0] = g4[1] = g4[2] = g4[3] = 0;
    g4[4] = g4[5] = g4[6] = g4[7] = 0;
    __builtin_amdgcn_tensor_load_to_lds(g0, g1, g2, g3, g4, 0);
    __builtin_amdgcn_s_wait_tensorcnt((short)0);
  }
  __syncthreads();

  const int lane = threadIdx.x & 31;
  const int wid  = threadIdx.x >> 5;                 // 0..7
  const int hi   = lane >> 4;                        // 0: lanes 0-15, 1: lanes 16-31
  const int l15  = lane & 15;

  const int m0   = m_blk + (wid << 5);               // 32 rows per wave
  const int col0 = n_blk + l15;
  const int col1 = col0 + 16;

  v8f c00, c01, c10, c11;
#pragma unroll
  for (int r = 0; r < 8; ++r) {
    int ma = m0 + r + 8 * hi;                        // first 16-row tile
    int mb = ma + 16;                                // second 16-row tile
    float bz0 = bias ? bias[col0] : 0.0f;
    float bz1 = bias ? bias[col1] : 0.0f;
    float v00 = bz0, v01 = bz1, v10 = bz0, v11 = bz1;
    if (resid) {
      v00 += resid[(size_t)ma * Nd + col0]; v01 += resid[(size_t)ma * Nd + col1];
      v10 += resid[(size_t)mb * Nd + col0]; v11 += resid[(size_t)mb * Nd + col1];
    }
    c00[r] = v00; c01[r] = v01; c10[r] = v10; c11[r] = v11;
  }

  union Frag { v16bf v; uint32_t u[8]; } a0, a1, b0, b1;
  const uint16_t* Arow0 = A + (size_t)(m0 + l15) * K;       // rows m0..m0+15
  const uint16_t* Arow1 = A + (size_t)(m0 + 16 + l15) * K;  // rows m0+16..m0+31
  const uint16_t* Bl0   = ldsB + (size_t)l15 * K;           // local col l15
  const uint16_t* Bl1   = ldsB + (size_t)(l15 + 16) * K;    // local col l15+16

  for (int k0 = 0; k0 < K; k0 += 32) {
#pragma unroll
    for (int i = 0; i < 8; ++i) {
      // 16-bit A 16x32 layout: lanes0-15 VGPR i -> K = {0,2,4,6,16,18,20,22}[i]; lanes16-31 +8
      int ka = k0 + 8 * hi + ((i < 4) ? (2 * i) : (16 + 2 * (i - 4)));
      a0.u[i] = *(const uint32_t*)(Arow0 + ka);
      a1.u[i] = *(const uint32_t*)(Arow1 + ka);
      // 16-bit B 32x16 layout (matches A): lanes0-15 VGPR i -> K = {0,2,4,6,16,18,20,22}[i]; lanes16-31 +8
      int kb = k0 + 8 * hi + ((i < 4) ? (2 * i) : (16 + 2 * (i - 4)));
      b0.u[i] = *(const uint32_t*)(Bl0 + kb);
      b1.u[i] = *(const uint32_t*)(Bl1 + kb);
    }
    c00 = __builtin_amdgcn_wmma_f32_16x16x32_bf16(false, a0.v, false, b0.v, (short)0, c00, false, false);
    c01 = __builtin_amdgcn_wmma_f32_16x16x32_bf16(false, a0.v, false, b1.v, (short)0, c01, false, false);
    c10 = __builtin_amdgcn_wmma_f32_16x16x32_bf16(false, a1.v, false, b0.v, (short)0, c10, false, false);
    c11 = __builtin_amdgcn_wmma_f32_16x16x32_bf16(false, a1.v, false, b1.v, (short)0, c11, false, false);
  }

#pragma unroll
  for (int r = 0; r < 8; ++r) {
    int ma = m0 + r + 8 * hi;
    int mb = ma + 16;
    float v00 = c00[r], v01 = c01[r], v10 = c10[r], v11 = c11[r];
    if (relu) {
      v00 = v00 > 0.f ? v00 : 0.f; v01 = v01 > 0.f ? v01 : 0.f;
      v10 = v10 > 0.f ? v10 : 0.f; v11 = v11 > 0.f ? v11 : 0.f;
    }
    size_t oa = (size_t)ma * Nd, ob = (size_t)mb * Nd;
    if (outF) {
      outF[oa + col0] = v00; outF[oa + col1] = v01;
      outF[ob + col0] = v10; outF[ob + col1] = v11;
    }
    if (outBf) {
      outBf[oa + col0] = f2bf(v00); outBf[oa + col1] = f2bf(v01);
      outBf[ob + col0] = f2bf(v10); outBf[ob + col1] = f2bf(v11);
    }
  }
}

// ---------------- deformable sampling ----------------
// One wave per (b,q,h); lane = channel of d_head. value f32 [NB,LENIN,256],
// off f32 [NB,LQ,192], logits f32 [NB,LQ,96] -> acc bf16 [NB,LQ,256]
__global__ void msw_sample(const float* __restrict__ value,
                           const float* __restrict__ off,
                           const float* __restrict__ logit,
                           uint16_t* __restrict__ acc_bf) {
  const int lane = threadIdx.x & 31;
  const int wave = (blockIdx.x * blockDim.x + threadIdx.x) >> 5;
  if (wave >= NB * LQ * NHEADS) return;
  const int h = wave % NHEADS;
  const int q = (wave / NHEADS) % LQ;
  const int b = wave / (NHEADS * LQ);

  const float rx = ((q & 127) + 0.5f) * (1.0f / 128.0f);
  const float ry = ((q >> 7)  + 0.5f) * (1.0f / 128.0f);

  // softmax over 12 (l,p) weights, redundantly per lane (broadcast loads)
  const float* lg = logit + ((size_t)(b * LQ + q)) * 96 + h * 12;
  float w[12]; float mx = -1e30f;
#pragma unroll
  for (int i = 0; i < 12; ++i) { w[i] = lg[i]; mx = fmaxf(mx, w[i]); }
  float s = 0.f;
#pragma unroll
  for (int i = 0; i < 12; ++i) { w[i] = __expf(w[i] - mx); s += w[i]; }
  const float inv = 1.0f / s;

  const int HW[3]  = {128, 64, 32};
  const int lsi[3] = {0, 16384, 20480};

  const float* offq = off + ((size_t)(b * LQ + q)) * 192 + h * 24; // h*24 + l*8 + p*2
  float acc = 0.f;
#pragma unroll
  for (int l = 0; l < NLEV; ++l) {
    const int W = HW[l], H = HW[l];
    const float* vbase = value + ((size_t)(b * LENIN + lsi[l])) * DMODEL + h * DHEAD + lane;
#pragma unroll
    for (int p = 0; p < NPTS; ++p) {
      float x = (rx + offq[l * 8 + p * 2 + 0] / (float)W) * W - 0.5f;
      float y = (ry + offq[l * 8 + p * 2 + 1] / (float)H) * H - 0.5f;
      float x0f = floorf(x), y0f = floorf(y);
      float dx = x - x0f,   dy = y - y0f;
      int x0 = (int)x0f, y0 = (int)y0f;
      float sp = 0.f;
#pragma unroll
      for (int cy = 0; cy < 2; ++cy) {
#pragma unroll
        for (int cx = 0; cx < 2; ++cx) {
          int ix = x0 + cx, iy = y0 + cy;
          float wc = (cx ? dx : 1.f - dx) * (cy ? dy : 1.f - dy);
          if (ix >= 0 && ix < W && iy >= 0 && iy < H)
            sp += wc * vbase[(size_t)(iy * W + ix) * DMODEL];   // 128B coalesced per corner
        }
      }
      acc += w[l * 4 + p] * inv * sp;
    }
  }
  acc_bf[((size_t)(b * LQ + q)) * DMODEL + h * DHEAD + lane] = f2bf(acc);
}

// ---------------- layernorm (wave per row of 256) ----------------
__global__ void msw_layernorm(const float* __restrict__ x, const float* __restrict__ gamma,
                              const float* __restrict__ beta, uint16_t* __restrict__ out) {
  const int lane = threadIdx.x & 31;
  const int row  = (blockIdx.x * blockDim.x + threadIdx.x) >> 5;
  if (row >= NB * LQ) return;
  const float* xr = x + (size_t)row * DMODEL;
  float v[8]; float s = 0.f, s2 = 0.f;
#pragma unroll
  for (int i = 0; i < 8; ++i) { v[i] = xr[lane + 32 * i]; s += v[i]; s2 += v[i] * v[i]; }
#pragma unroll
  for (int m = 16; m > 0; m >>= 1) { s += __shfl_xor(s, m, 32); s2 += __shfl_xor(s2, m, 32); }
  const float mu  = s * (1.f / DMODEL);
  const float var = s2 * (1.f / DMODEL) - mu * mu;
  const float r   = rsqrtf(var + 1e-5f);
  uint16_t* o = out + (size_t)row * DMODEL;
#pragma unroll
  for (int i = 0; i < 8; ++i) {
    int ch = lane + 32 * i;
    o[ch] = f2bf((v[i] - mu) * r * gamma[ch] + beta[ch]);
  }
}

// ---------------- host ----------------
extern "C" void kernel_launch(void* const* d_in, const int* in_sizes, int n_in,
                              void* d_out, int out_size, void* d_ws, size_t ws_size,
                              hipStream_t stream) {
  const float* src     = (const float*)d_in[0];
  const float* flow    = (const float*)d_in[1];
  const float* w_value = (const float*)d_in[4];
  const float* b_value = (const float*)d_in[5];
  const float* w_off   = (const float*)d_in[6];
  const float* b_off   = (const float*)d_in[7];
  const float* w_attn  = (const float*)d_in[8];
  const float* b_attn  = (const float*)d_in[9];
  const float* w_out   = (const float*)d_in[10];
  const float* b_out   = (const float*)d_in[11];
  const float* gamma   = (const float*)d_in[12];
  const float* beta    = (const float*)d_in[13];
  const float* w1      = (const float*)d_in[14];
  const float* b1      = (const float*)d_in[15];
  const float* w2      = (const float*)d_in[16];
  const float* b2      = (const float*)d_in[17];

  uint8_t* ws = (uint8_t*)d_ws;
  size_t cur = 0;
  auto alloc = [&](size_t bytes) -> void* {
    void* p = ws + cur;
    cur += (bytes + 255) & ~(size_t)255;
    return p;
  };

  const int M_v = NB * LENIN;   // 43008 (= 168 * 256)
  const int M_q = NB * LQ;      // 32768 (= 128 * 256)

  uint16_t* src_bf  = (uint16_t*)alloc((size_t)M_v * DMODEL * 2);
  uint16_t* flow_bf = (uint16_t*)alloc((size_t)M_q * DFLOW * 2);
  uint16_t* wv_t    = (uint16_t*)alloc((size_t)DMODEL * DMODEL * 2);
  uint16_t* woff_t  = (uint16_t*)alloc((size_t)192 * DFLOW * 2);
  uint16_t* wattn_t = (uint16_t*)alloc((size_t)96 * DFLOW * 2);
  uint16_t* wout_t  = (uint16_t*)alloc((size_t)DMODEL * DMODEL * 2);
  uint16_t* w1_t    = (uint16_t*)alloc((size_t)DFFN * DMODEL * 2);
  uint16_t* w2_t    = (uint16_t*)alloc((size_t)DMODEL * DFFN * 2);
  float*    value   = (float*)   alloc((size_t)M_v * DMODEL * 4);
  float*    offb    = (float*)   alloc((size_t)M_q * 192 * 4);
  float*    logit   = (float*)   alloc((size_t)M_q * 96 * 4);
  uint16_t* acc_bf  = (uint16_t*)alloc((size_t)M_q * DMODEL * 2);
  float*    src2    = (float*)   alloc((size_t)M_q * DMODEL * 4);
  uint16_t* ln_bf   = (uint16_t*)alloc((size_t)M_q * DMODEL * 2);
  uint16_t* h_bf    = (uint16_t*)alloc((size_t)M_q * DFFN * 2);

  auto cvt = [&](const float* in, uint16_t* out, int n) {
    msw_f32_to_bf16<<<(n + 255) / 256, 256, 0, stream>>>(in, out, n);
  };
  auto trans = [&](const float* W, uint16_t* Wt, int K, int Nd) {
    int n = K * Nd;
    msw_transpose_bf16<<<(n + 255) / 256, 256, 0, stream>>>(W, Wt, K, Nd);
  };
  auto gemm = [&](const uint16_t* A, const uint16_t* Bt, const float* bias,
                  const float* resid, float* outF, uint16_t* outBf,
                  int M, int K, int Nd, int relu) {
    int blocks = (M >> 8) * (Nd >> 5);               // 256-row x 32-col tiles
    size_t ldsBytes = (size_t)32 * K * 2;            // B slab
    msw_gemm_bf16<<<blocks, 256, ldsBytes, stream>>>(A, Bt, bias, resid, outF, outBf, M, K, Nd, relu);
  };

  // stage 0: precision conversion
  cvt(src,  src_bf,  M_v * DMODEL);
  cvt(flow, flow_bf, M_q * DFLOW);
  trans(w_value, wv_t,    DMODEL, DMODEL);
  trans(w_off,   woff_t,  DFLOW, 192);
  trans(w_attn,  wattn_t, DFLOW, 96);
  trans(w_out,   wout_t,  DMODEL, DMODEL);
  trans(w1,      w1_t,    DMODEL, DFFN);
  trans(w2,      w2_t,    DFFN, DMODEL);

  // stage 1: projections
  gemm(src_bf,  wv_t,    b_value, nullptr, value, nullptr, M_v, DMODEL, DMODEL, 0);
  gemm(flow_bf, woff_t,  b_off,   nullptr, offb,  nullptr, M_q, DFLOW, 192, 0);
  gemm(flow_bf, wattn_t, b_attn,  nullptr, logit, nullptr, M_q, DFLOW, 96, 0);

  // stage 2: softmax + deformable bilinear sampling
  {
    int waves = NB * LQ * NHEADS;
    msw_sample<<<(waves * 32 + 255) / 256, 256, 0, stream>>>(value, offb, logit, acc_bf);
  }

  // stage 3: output projection (src2 = acc @ w_out + b_out)
  gemm(acc_bf, wout_t, b_out, nullptr, src2, nullptr, M_q, DMODEL, DMODEL, 0);

  // stage 4: layernorm -> bf16
  {
    int waves = M_q;
    msw_layernorm<<<(waves * 32 + 255) / 256, 256, 0, stream>>>(src2, gamma, beta, ln_bf);
  }

  // stage 5: FFN with fused residual: out = src2 + relu(ln@w1+b1)@w2 + b2
  gemm(ln_bf, w1_t, b1, nullptr, nullptr, h_bf, M_q, DMODEL, DFFN, 1);
  gemm(h_bf,  w2_t, b2, src2, (float*)d_out, nullptr, M_q, DFFN, DMODEL, 0);
}
